// InteractingLayer_47802986004579
// MI455X (gfx1250) — compile-verified, matrix operand order checked by reading, not run on hardware
//
#include <hip/hip_runtime.h>
#include <hip/hip_bf16.h>

typedef __bf16 bf16_t;
typedef __attribute__((ext_vector_type(16))) __bf16 v16bf;
typedef __attribute__((ext_vector_type(4)))  __bf16 v4bf;
typedef __attribute__((ext_vector_type(8)))  float  v8f;

#define BATCH 4096
#define FNUM  50      // fields
#define EMB   64      // embedding size (K of projections)
#define DH    128     // D*H
#define HEADS 4
#define HD    32      // per-head dim
#define FP    64      // padded field count (4 tiles of 16)

// LDS layout offsets (bytes).  Total 152 KB -> 2 blocks per WGP (320 KB pool).
#define OFF_X    0                     // bf16 [64][64]            8192
#define OFF_QT   8192                  // bf16 [4][32][64]        16384  (Q^T)
#define OFF_KT   24576                 // bf16 [4][32][64]        16384  (K^T)
#define OFF_VT   40960                 // bf16 [4][32][64]        16384  (V^T)
#define OFF_RES  57344                 // f32  [64][128]          32768
#define OFF_S    90112                 // f32  [4][64][64]        65536; bf16 P overlays in place
#define SMEM_BYTES 155648

// A/B fragment from a row-contiguous store (row r holds its K-values
// contiguously).  k0 in {0,32}.
__device__ __forceinline__ v16bf load_frag16(const bf16_t* __restrict__ buf,
                                             int rbase, int ld, int k0, int lane) {
  const bf16_t* p = buf + (rbase + (lane & 15)) * ld + k0 + ((lane >> 4) << 3);
  v16bf f;
#pragma unroll
  for (int j = 0; j < 8; ++j) { f[j] = p[j]; f[j + 8] = p[j + 16]; }
  return f;
}

// A/B fragment gather from a transposed store T[k][r] (ldt = row length in r).
__device__ __forceinline__ v16bf load_frag16_col(const bf16_t* __restrict__ buf,
                                                 int rbase, int ldt, int k0, int lane) {
  const int r  = rbase + (lane & 15);
  const int kk = k0 + ((lane >> 4) << 3);
  v16bf f;
#pragma unroll
  for (int j = 0; j < 8; ++j) {
    f[j]     = buf[(kk + j) * ldt + r];
    f[j + 8] = buf[(kk + 16 + j) * ldt + r];
  }
  return f;
}

__device__ __forceinline__ v8f wmma_bf16(v16bf a, v16bf b, v8f c) {
  return __builtin_amdgcn_wmma_f32_16x16x32_bf16(false, a, false, b,
                                                 (short)0, c, false, false);
}

// Convert the four [E=64][DH=128] fp32 weights to bf16, transposed to
// Wt[w][n][k] so WMMA B-fragments are contiguous.  64 KB total -> L2 resident.
__global__ void prep_weights_kernel(const float* __restrict__ Wq,
                                    const float* __restrict__ Wk,
                                    const float* __restrict__ Wv,
                                    const float* __restrict__ Wr,
                                    bf16_t* __restrict__ Wt) {
  int idx = blockIdx.x * 256 + threadIdx.x;
  const int per = DH * EMB;                     // 8192
  if (idx >= 4 * per) return;
  int w = idx / per, rem = idx - w * per;
  int n = rem >> 6, k = rem & 63;               // n in [0,128), k in [0,64)
  const float* W = (w == 0) ? Wq : (w == 1) ? Wk : (w == 2) ? Wv : Wr;
  Wt[idx] = (bf16_t)W[k * DH + n];
}

__global__ __launch_bounds__(128)
void interacting_layer_kernel(const float* __restrict__ X,
                              const bf16_t* __restrict__ Wt,
                              float* __restrict__ out) {
  extern __shared__ char smem[];
  bf16_t* sX   = (bf16_t*)(smem + OFF_X);
  bf16_t* sQt  = (bf16_t*)(smem + OFF_QT);
  bf16_t* sKt  = (bf16_t*)(smem + OFF_KT);
  bf16_t* sVt  = (bf16_t*)(smem + OFF_VT);
  float*  sRes = (float*) (smem + OFF_RES);
  float*  sS   = (float*) (smem + OFF_S);

  const int b    = blockIdx.x;
  const int tid  = threadIdx.x;
  const int w    = __builtin_amdgcn_readfirstlane(tid >> 5);  // wave-uniform SGPR
  const int lane = tid & 31;

  // ---- Stage X_b into LDS as bf16 (float4 loads, b64 stores), pad rows 0 ---
  const float* xb = X + (size_t)b * FNUM * EMB;
  for (int i4 = tid; i4 < FP * EMB / 4; i4 += 128) {
    const int r = i4 >> 4, c = (i4 & 15) * 4;
    float4 v = make_float4(0.f, 0.f, 0.f, 0.f);
    if (r < FNUM) v = *(const float4*)(xb + r * EMB + c);
    v4bf p = { (bf16_t)v.x, (bf16_t)v.y, (bf16_t)v.z, (bf16_t)v.w };
    *(v4bf*)(sX + r * EMB + c) = p;
  }
  __syncthreads();

  // ---- Phase 1: wave w computes X @ W_w  (0=Q 1=K 2=V 3=residual) ----------
  // X A-fragments hoisted once into registers (64 VGPRs).
  v16bf xa[4][2];
#pragma unroll
  for (int mt = 0; mt < 4; ++mt) {
    xa[mt][0] = load_frag16(sX, mt * 16, EMB, 0,  lane);
    xa[mt][1] = load_frag16(sX, mt * 16, EMB, 32, lane);
  }
  const bf16_t* Ww = Wt + (size_t)w * DH * EMB;   // Wt[n][k], ld = 64
  if (w < 3) {                                    // scalar branch
    bf16_t* dstT = (w == 0) ? sQt : (w == 1) ? sKt : sVt;  // [h][d][f]
    for (int nt = 0; nt < 8; ++nt) {
      v16bf b0 = load_frag16(Ww, nt * 16, EMB, 0,  lane);
      v16bf b1 = load_frag16(Ww, nt * 16, EMB, 32, lane);
#pragma unroll
      for (int mt = 0; mt < 4; ++mt) {
        v8f acc = {};
        acc = wmma_bf16(xa[mt][0], b0, acc);
        acc = wmma_bf16(xa[mt][1], b1, acc);
        const int n = nt * 16 + (lane & 15);         // column in [0,128)
        const int m = mt * 16 + ((lane >> 4) << 3);  // row base
        bf16_t* dst = dstT + (n >> 5) * 2048 + (n & 31) * FP + m;  // contiguous
#pragma unroll
        for (int v = 0; v < 8; ++v) dst[v] = (bf16_t)acc[v];       // -> b128
      }
    }
  } else {                                        // residual projection, fp32
    for (int nt = 0; nt < 8; ++nt) {
      v16bf b0 = load_frag16(Ww, nt * 16, EMB, 0,  lane);
      v16bf b1 = load_frag16(Ww, nt * 16, EMB, 32, lane);
#pragma unroll
      for (int mt = 0; mt < 4; ++mt) {
        v8f acc = {};
        acc = wmma_bf16(xa[mt][0], b0, acc);
        acc = wmma_bf16(xa[mt][1], b1, acc);
        const int n = nt * 16 + (lane & 15);
        const int m = mt * 16 + ((lane >> 4) << 3);
#pragma unroll
        for (int v = 0; v < 8; ++v) sRes[(m + v) * DH + n] = acc[v];
      }
    }
  }
  __syncthreads();

  // ---- Phase 2: wave h = w owns head h -------------------------------------
  const int h = w;
  const bf16_t* Qt = sQt + h * 2048;   // [32][64]  Qt[d][f]
  const bf16_t* Kt = sKt + h * 2048;   // [32][64]  Kt[d][g]
  const bf16_t* Vt = sVt + h * 2048;   // [32][64]  Vt[d][g]
  float*  Sh = sS + h * 4096;          // fp32 [64][64] scores
  bf16_t* Pb = (bf16_t*)Sh;            // bf16 overlay, row stride 128

  // scores = Q K^T : K-dim = 32, one WMMA step; 4x4 tiles.  B frags hoisted.
  {
    v16bf kb[4];
#pragma unroll
    for (int nt = 0; nt < 4; ++nt) kb[nt] = load_frag16_col(Kt, nt * 16, FP, 0, lane);
    for (int mt = 0; mt < 4; ++mt) {
      v16bf a = load_frag16_col(Qt, mt * 16, FP, 0, lane);
#pragma unroll
      for (int nt = 0; nt < 4; ++nt) {
        v8f acc = {};
        acc = wmma_bf16(a, kb[nt], acc);
        const int g = nt * 16 + (lane & 15);
        const int m = mt * 16 + ((lane >> 4) << 3);
#pragma unroll
        for (int v = 0; v < 8; ++v) Sh[(m + v) * FP + g] = acc[v];
      }
    }
  }

  // softmax over g<50 per row, fp32, register-staged; bf16 probs written in
  // place over the score row (compiler barrier orders the overlay).
  for (int rr = 0; rr < 2; ++rr) {
    const int f = lane + rr * 32;
    bf16_t* prow = Pb + f * 128;
    if (f < FNUM) {
      const float* srow = Sh + f * FP;
      float e[FNUM];
#pragma unroll
      for (int g = 0; g < FNUM; ++g) e[g] = srow[g];
      float mx = -3.0e38f;
#pragma unroll
      for (int g = 0; g < FNUM; ++g) mx = fmaxf(mx, e[g]);
      float sum = 0.0f;
#pragma unroll
      for (int g = 0; g < FNUM; ++g) { e[g] = __expf(e[g] - mx); sum += e[g]; }
      const float inv = 1.0f / sum;
      asm volatile("" ::: "memory");    // order bf16 overlay after fp32 reads
#pragma unroll
      for (int g = 0; g < FNUM; ++g) prow[g] = (bf16_t)(e[g] * inv);
#pragma unroll
      for (int g = FNUM; g < FP; ++g) prow[g] = (bf16_t)0.0f;
    } else {
#pragma unroll
      for (int g = 0; g < FP; ++g) prow[g] = (bf16_t)0.0f;
    }
  }
  // wave-local LDS RAW: hardware DScnt ordering (compiler-inserted waits).

  // out_h = P @ V (+ residual, ReLU): 4 M-tiles x 2 N-tiles x 2 K-steps.
  {
    v16bf vb[2][2];
#pragma unroll
    for (int nt = 0; nt < 2; ++nt) {
      vb[nt][0] = load_frag16(Vt, nt * 16, FP, 0,  lane);
      vb[nt][1] = load_frag16(Vt, nt * 16, FP, 32, lane);
    }
    float* ob = out + (size_t)b * FNUM * DH;
    for (int mt = 0; mt < 4; ++mt) {
      v16bf a0 = load_frag16(Pb, mt * 16, 128, 0,  lane);  // overlay rows, ld=128
      v16bf a1 = load_frag16(Pb, mt * 16, 128, 32, lane);
#pragma unroll
      for (int nt = 0; nt < 2; ++nt) {
        v8f acc = {};
        acc = wmma_bf16(a0, vb[nt][0], acc);
        acc = wmma_bf16(a1, vb[nt][1], acc);
        const int d   = nt * 16 + (lane & 15);
        const int m   = mt * 16 + ((lane >> 4) << 3);
        const int col = h * HD + d;
#pragma unroll
        for (int v = 0; v < 8; ++v) {
          const int f = m + v;
          if (f < FNUM) {
            float val = acc[v] + sRes[f * DH + col];
            ob[f * DH + col] = fmaxf(val, 0.0f);
          }
        }
      }
    }
  }
}

extern "C" void kernel_launch(void* const* d_in, const int* in_sizes, int n_in,
                              void* d_out, int out_size, void* d_ws, size_t ws_size,
                              hipStream_t stream) {
  const float* X  = (const float*)d_in[0];
  const float* Wq = (const float*)d_in[1];
  const float* Wk = (const float*)d_in[2];
  const float* Wv = (const float*)d_in[3];
  const float* Wr = (const float*)d_in[4];
  bf16_t* Wt = (bf16_t*)d_ws;                       // 4*128*64 bf16 = 64 KB

  prep_weights_kernel<<<(4 * DH * EMB + 255) / 256, 256, 0, stream>>>(Wq, Wk, Wv, Wr, Wt);
  interacting_layer_kernel<<<BATCH, 128, SMEM_BYTES, stream>>>(X, Wt, (float*)d_out);
}
